// NewLLMLayerwise_54168127537687
// MI455X (gfx1250) — compile-verified, matrix operand order checked by reading, not run on hardware
//
#include <hip/hip_runtime.h>
#include <hip/hip_bf16.h>
#include <cstdint>

// ---------------------------------------------------------------------------
// NewLLMLayerwise forward for MI455X (gfx1250, wave32, WMMA + TDM).
// fp32 weights -> bf16 once per launch; all GEMMs via v_wmma_f32_16x16x32_bf16
// (fp32 accumulate); gate GEMMs fused (N=3072, mixed activation); A tiles
// staged by the Tensor Data Mover when available; LN/context math in fp32.
// ---------------------------------------------------------------------------

typedef unsigned short ushort_t;
typedef __attribute__((ext_vector_type(16))) __bf16 v16bf;
typedef __attribute__((ext_vector_type(8)))  float  v8f;
typedef __attribute__((ext_vector_type(4)))  int    v4i;

#define V_   32000
#define E_   1024
#define C_   1024
#define H_   4096
#define L_   6
#define N_   4096      // B*S tokens
#define G3_  (3 * C_)  // fused gates width
#define EPS_ 1e-5f

#define BM 128
#define BN 256
#define BK 32
#define LDK 40         // padded K-stride in LDS (elements): 64B row + 16B pad

#if __has_builtin(__builtin_amdgcn_tensor_load_to_lds) && __has_builtin(__builtin_amdgcn_s_wait_tensorcnt)
#define USE_TDM 1
#else
#define USE_TDM 0
#endif

__device__ __forceinline__ ushort_t f32_to_bf16(float f) {
  unsigned u = __float_as_uint(f);
  unsigned r = u + 0x7FFFu + ((u >> 16) & 1u);   // round-to-nearest-even
  return (ushort_t)(r >> 16);
}

union FragU { v16bf v; v4i i[2]; };
union Pack8 { v4i v; ushort_t h[8]; };

#if USE_TDM
typedef __attribute__((ext_vector_type(4))) unsigned int v4u;
typedef __attribute__((ext_vector_type(8))) int v8i;
// TDM: DMA a 128-row x 32-col bf16 tile (row stride = strideElems) from
// global into LDS, inserting 16B of pad after every 64B so the LDS layout
// matches the LDK=40 padded rows the WMMA fragment loads expect.
__device__ __forceinline__ void tdm_load_tile_a(const ushort_t* g,
                                                const ushort_t* lds,
                                                int strideElems) {
  unsigned long long ga = (unsigned long long)(uintptr_t)g;
  unsigned la = (unsigned)(uintptr_t)lds;   // low 32 bits of flat = LDS offset
  v4u g0;
  g0.x = 1u;                                // count=1, user mode, no gather
  g0.y = la;                                // lds_addr (bytes)
  g0.z = (unsigned)ga;                      // global_addr[31:0]
  g0.w = (unsigned)((ga >> 32) & 0x01FFFFFFu) | (2u << 30);  // addr[56:32], type=2
  v8i g1;
  g1[0] = (int)((1u << 16)      // data_size = 2 bytes
              | (1u << 20)      // pad_enable
              | (3u << 22)      // pad_interval = 16 DWORDs (64B)
              | (3u << 25));    // pad_amount   = 4 DWORDs  (16B)
  g1[1] = (int)(32u << 16);     // tensor_dim0 = 32 (lo16 in [31:16])
  g1[2] = (int)(128u << 16);    // tensor_dim1 = 128 (lo16 in [31:16])
  g1[3] = (int)(32u << 16);     // tile_dim0 = 32
  g1[4] = (int)128u;            // tile_dim1 = 128, tile_dim2 = 0 (2D)
  g1[5] = (int)strideElems;     // tensor_dim0_stride lo32
  g1[6] = 0;                    // stride0 hi16 = 0, stride1 lo16 = 0
  g1[7] = 0;
  v4i z = (v4i)0;
#if __clang_major__ >= 23
  v8i z8 = (v8i)0;
  __builtin_amdgcn_tensor_load_to_lds(g0, g1, z, z, z8, 0);
#else
  __builtin_amdgcn_tensor_load_to_lds(g0, g1, z, z, 0);
#endif
}
#endif

// ------------------------------ fp32 -> bf16 -------------------------------
__global__ void cvt_bf16_kernel(const float* __restrict__ src,
                                ushort_t* __restrict__ dst, long n) {
  long i = (long)blockIdx.x * blockDim.x + threadIdx.x;
  long stride = (long)gridDim.x * blockDim.x;
  for (; i < n; i += stride) dst[i] = f32_to_bf16(src[i]);
}

// ------------- pack Wd|Wf|Wi -> bf16 [L][H][3C] column-concat --------------
__global__ void cvt_gates_kernel(const float* __restrict__ Wd,
                                 const float* __restrict__ Wf,
                                 const float* __restrict__ Wi,
                                 ushort_t* __restrict__ dst) {
  long i = (long)blockIdx.x * blockDim.x + threadIdx.x;
  long stride = (long)gridDim.x * blockDim.x;
  const long total = (long)L_ * H_ * C_;
  for (; i < total; i += stride) {
    long lk = i / C_;            // l*H + k
    int  c  = (int)(i - lk * C_);
    long base = lk * G3_;
    dst[base + c]          = f32_to_bf16(Wd[i]);
    dst[base + C_ + c]     = f32_to_bf16(Wf[i]);
    dst[base + 2 * C_ + c] = f32_to_bf16(Wi[i]);
  }
}

__global__ void pack_bias_kernel(const float* __restrict__ bd,
                                 const float* __restrict__ bfg,
                                 const float* __restrict__ bi,
                                 float* __restrict__ dst) {
  int i = blockIdx.x * blockDim.x + threadIdx.x;
  if (i < L_ * C_) {
    int l = i / C_, c = i - l * C_;
    dst[l * G3_ + c]          = bd[i];
    dst[l * G3_ + C_ + c]     = bfg[i];
    dst[l * G3_ + 2 * C_ + c] = bi[i];
  }
}

// ------------------------- zero context + concat half ----------------------
__global__ void init_ctx_kernel(float* __restrict__ ctx, ushort_t* __restrict__ fnn) {
  long i = (long)blockIdx.x * blockDim.x + threadIdx.x;
  long stride = (long)gridDim.x * blockDim.x;
  for (; i < (long)N_ * C_; i += stride) {
    ctx[i] = 0.0f;
    long tok = i / C_, e = i - tok * C_;
    fnn[tok * (E_ + C_) + E_ + e] = 0;  // bf16 zero
  }
}

// ------------------- embedding gather + LayerNorm -> bf16 x ----------------
__global__ __launch_bounds__(256)
void embed_ln_kernel(const int* __restrict__ ids, const float* __restrict__ emb,
                     const float* __restrict__ g, const float* __restrict__ b,
                     ushort_t* __restrict__ fnn) {
  int tok = blockIdx.x, t = threadIdx.x;
  const float* row = emb + (long)ids[tok] * E_;
  float vals[4], s = 0.f, sq = 0.f;
#pragma unroll
  for (int i = 0; i < 4; ++i) {
    int e = t + i * 256;
    float v = row[e];
    vals[i] = v; s += v; sq += v * v;
  }
  __shared__ float rs[256], rq[256];
  rs[t] = s; rq[t] = sq; __syncthreads();
  for (int o = 128; o > 0; o >>= 1) {
    if (t < o) { rs[t] += rs[t + o]; rq[t] += rq[t + o]; }
    __syncthreads();
  }
  float mean = rs[0] * (1.f / E_);
  float var  = rq[0] * (1.f / E_) - mean * mean;
  float inv  = rsqrtf(var + EPS_);
#pragma unroll
  for (int i = 0; i < 4; ++i) {
    int e = t + i * 256;
    float o = (vals[i] - mean) * inv * g[e] + b[e];
    fnn[(long)tok * (E_ + C_) + e] = f32_to_bf16(o);
  }
}

// -------- context update: c = clip(LN(f*c + i*d)); gates packed [N,3C] -----
__global__ __launch_bounds__(256)
void ctx_ln_kernel(const float* __restrict__ gates, float* __restrict__ ctx,
                   const float* __restrict__ g, const float* __restrict__ b,
                   ushort_t* __restrict__ fnn) {
  int tok = blockIdx.x, t = threadIdx.x;
  long cbase = (long)tok * C_;
  long gbase = (long)tok * G3_;
  float u[4], s = 0.f, sq = 0.f;
#pragma unroll
  for (int i = 0; i < 4; ++i) {
    int e = t + i * 256;
    float d  = gates[gbase + e];
    float fg = gates[gbase + C_ + e];
    float ig = gates[gbase + 2 * C_ + e];
    float v = fg * ctx[cbase + e] + ig * d;
    u[i] = v; s += v; sq += v * v;
  }
  __shared__ float rs[256], rq[256];
  rs[t] = s; rq[t] = sq; __syncthreads();
  for (int o = 128; o > 0; o >>= 1) {
    if (t < o) { rs[t] += rs[t + o]; rq[t] += rq[t + o]; }
    __syncthreads();
  }
  float mean = rs[0] * (1.f / C_);
  float var  = rq[0] * (1.f / C_) - mean * mean;
  float inv  = rsqrtf(var + EPS_);
#pragma unroll
  for (int i = 0; i < 4; ++i) {
    int e = t + i * 256;
    float v = (u[i] - mean) * inv * g[e] + b[e];
    v = fminf(fmaxf(v, -10.f), 10.f);
    ctx[cbase + e] = v;
    fnn[(long)tok * (E_ + C_) + E_ + e] = f32_to_bf16(v);
  }
}

// --------------------------- bf16 WMMA GEMM --------------------------------
// C[M,N] = act(A[M,K] @ B[K,N] + bias[N]); A,B bf16 row-major; fp32 accum.
// Block tile 128x256, K-step 32; 8 wave32 waves as 2x4; wave tile 64x64.
// A tile: TDM DMA (padded rows) when available, else reg staging.
// B tile: reg load + LDS scatter-transpose so both A and B fragment loads are
// two aligned ds_load_b128 per lane (ISA 7.12.2 16-bit layouts).
// ACT: 0 none, 1 relu, 4 gates (col<C_: tanh else sigmoid). OUTBF: bf16 out.
template <int ACT, int OUTBF>
__global__ __launch_bounds__(256)
void gemm_bf16_kernel(const ushort_t* __restrict__ A, const ushort_t* __restrict__ Bw,
                      const float* __restrict__ bias,
                      float* __restrict__ outF, ushort_t* __restrict__ outB,
                      int M, int N, int K) {
  __shared__ __align__(16) ushort_t ldsA[2][BM * LDK];
  __shared__ __align__(16) ushort_t ldsB[2][BN * LDK];

  const int t     = threadIdx.x;
  const int lane  = t & 31;
  const int wave  = t >> 5;
  const int waveM = wave >> 2;   // 0..1  (64 rows each)
  const int waveN = wave & 3;    // 0..3  (64 cols each)
  const int mBase = blockIdx.y * BM;
  const int nBase = blockIdx.x * BN;

  v8f acc[4][4];
#pragma unroll
  for (int mt = 0; mt < 4; ++mt)
#pragma unroll
    for (int nt = 0; nt < 4; ++nt) acc[mt][nt] = (v8f)0.0f;

  const int KT = K / BK;

  // ---- prologue: stage tile 0 ----
#if USE_TDM
  if (wave == 0) tdm_load_tile_a(A + (size_t)mBase * K, &ldsA[0][0], K);
#else
#pragma unroll
  for (int p = 0; p < 2; ++p) {
    int flat = p * 256 + t;
    int row = flat >> 2, ch = flat & 3;               // 4 x 16B per A row
    v4i v = *(const v4i*)(A + (long)(mBase + row) * K + ch * 8);
    *(v4i*)(&ldsA[0][row * LDK + ch * 8]) = v;
  }
#endif
#pragma unroll
  for (int p = 0; p < 4; ++p) {
    int flat = p * 256 + t;
    int k = flat >> 5, nc = flat & 31;                // 32 x 16B per B row
    Pack8 u; u.v = *(const v4i*)(Bw + (long)k * N + nBase + nc * 8);
#pragma unroll
    for (int i = 0; i < 8; ++i) ldsB[0][(nc * 8 + i) * LDK + k] = u.h[i];
  }
#if USE_TDM
  if (wave == 0) __builtin_amdgcn_s_wait_tensorcnt(0);
#endif
  __syncthreads();

  const int k0  = (lane < 16) ? 0 : 8;
  const int r16 = lane & 15;

  for (int kt = 0; kt < KT; ++kt) {
    const int cur = kt & 1, nxt = cur ^ 1;
    const bool pre = (kt + 1) < KT;

    // ---- prefetch next tile (overlaps WMMA below) ----
    Pack8 bReg[4];
    int bk[4], bnc[4];
#if !USE_TDM
    v4i aReg[2]; int arow[2], ach[2];
#endif
    if (pre) {
      const long kOff = (long)(kt + 1) * BK;
#if USE_TDM
      if (wave == 0)
        tdm_load_tile_a(A + (size_t)mBase * K + kOff, &ldsA[nxt][0], K);
#else
#pragma unroll
      for (int p = 0; p < 2; ++p) {
        int flat = p * 256 + t;
        arow[p] = flat >> 2; ach[p] = flat & 3;
        aReg[p] = *(const v4i*)(A + (long)(mBase + arow[p]) * K + kOff + ach[p] * 8);
      }
#endif
#pragma unroll
      for (int p = 0; p < 4; ++p) {
        int flat = p * 256 + t;
        bk[p] = flat >> 5; bnc[p] = flat & 31;
        bReg[p].v = *(const v4i*)(Bw + (kOff + bk[p]) * N + nBase + bnc[p] * 8);
      }
    }
    // deep prefetch (kt+2) into GL2
    if (kt + 2 < KT) {
      const long k2 = (long)(kt + 2) * BK;
      __builtin_prefetch(Bw + (k2 + (t & 31)) * N + nBase + ((t >> 5) * 32), 0, 1);
      __builtin_prefetch(A + (long)(mBase + (t >> 1)) * K + k2 + ((t & 1) * 16), 0, 1);
    }

    // ---- B fragments (persist across mt), then per-mt A fragment ----
    FragU bfr[4];
#pragma unroll
    for (int nt = 0; nt < 4; ++nt) {
      const ushort_t* base = &ldsB[cur][(waveN * 64 + nt * 16 + r16) * LDK + k0];
      bfr[nt].i[0] = *(const v4i*)(base);
      bfr[nt].i[1] = *(const v4i*)(base + 16);
    }
#pragma unroll
    for (int mt = 0; mt < 4; ++mt) {
      FragU afr;
      const ushort_t* base = &ldsA[cur][(waveM * 64 + mt * 16 + r16) * LDK + k0];
      afr.i[0] = *(const v4i*)(base);
      afr.i[1] = *(const v4i*)(base + 16);
#pragma unroll
      for (int nt = 0; nt < 4; ++nt)
        acc[mt][nt] = __builtin_amdgcn_wmma_f32_16x16x32_bf16(
            false, afr.v, false, bfr[nt].v, (short)0, acc[mt][nt],
            false, false);
    }

    // ---- commit prefetch into the other LDS buffer ----
    if (pre) {
#if !USE_TDM
#pragma unroll
      for (int p = 0; p < 2; ++p)
        *(v4i*)(&ldsA[nxt][arow[p] * LDK + ach[p] * 8]) = aReg[p];
#endif
#pragma unroll
      for (int p = 0; p < 4; ++p)
#pragma unroll
        for (int i = 0; i < 8; ++i)
          ldsB[nxt][(bnc[p] * 8 + i) * LDK + bk[p]] = bReg[p].h[i];
    }
#if USE_TDM
    if (pre && wave == 0) __builtin_amdgcn_s_wait_tensorcnt(0);
#endif
    __syncthreads();
  }

  // ---- epilogue: bias + activation, write bf16 or f32 ----
#pragma unroll
  for (int mt = 0; mt < 4; ++mt) {
#pragma unroll
    for (int nt = 0; nt < 4; ++nt) {
      const int col = nBase + waveN * 64 + nt * 16 + r16;
      const float bv = bias[col];
      const int rowBase = mBase + waveM * 64 + mt * 16 + ((lane >> 4) << 3);
#pragma unroll
      for (int r = 0; r < 8; ++r) {
        float v = acc[mt][nt][r] + bv;
        if (ACT == 1) v = fmaxf(v, 0.f);
        else if (ACT == 4) v = (col < C_) ? tanhf(v) : (1.f / (1.f + expf(-v)));
        const long idx = (long)(rowBase + r) * N + col;
        if (OUTBF) outB[idx] = f32_to_bf16(v);
        else       outF[idx] = v;
      }
    }
  }
}

// ---------------------------------------------------------------------------
extern "C" void kernel_launch(void* const* d_in, const int* in_sizes, int n_in,
                              void* d_out, int out_size, void* d_ws, size_t ws_size,
                              hipStream_t stream) {
  (void)in_sizes; (void)n_in; (void)out_size;
  const int*   ids  = (const int*)  d_in[0];
  const float* emb  = (const float*)d_in[1];
  const float* en_g = (const float*)d_in[2];
  const float* en_b = (const float*)d_in[3];
  const float* W1   = (const float*)d_in[4];
  const float* b1   = (const float*)d_in[5];
  const float* Wd   = (const float*)d_in[6];
  const float* bd   = (const float*)d_in[7];
  const float* Wf   = (const float*)d_in[8];
  const float* bf_  = (const float*)d_in[9];
  const float* Wi   = (const float*)d_in[10];
  const float* bi   = (const float*)d_in[11];
  const float* ln_g = (const float*)d_in[12];
  const float* ln_b = (const float*)d_in[13];
  const float* Wout = (const float*)d_in[14];
  const float* bout = (const float*)d_in[15];
  float* out = (float*)d_out;

  // ---- workspace carve-up (~630 MB) ----
  char* ws = (char*)d_ws;
  size_t off = 0;
  auto carve = [&](size_t bytes) -> void* {
    void* p = ws + off;
    off += (bytes + 255) & ~(size_t)255;
    return p;
  };
  ushort_t* W1b   = (ushort_t*)carve((size_t)L_ * (E_ + C_) * H_ * 2);
  ushort_t* Wgb   = (ushort_t*)carve((size_t)L_ * H_ * G3_ * 2);      // Wd|Wf|Wi
  ushort_t* Woutb = (ushort_t*)carve((size_t)H_ * V_ * 2);
  ushort_t* fnnB  = (ushort_t*)carve((size_t)N_ * (E_ + C_) * 2);
  ushort_t* hidB  = (ushort_t*)carve((size_t)N_ * H_ * 2);
  float*    gatesF= (float*)   carve((size_t)N_ * G3_ * 4);
  float*    ctxF  = (float*)   carve((size_t)N_ * C_ * 4);
  float*    bcat  = (float*)   carve((size_t)L_ * G3_ * 4);
  if (ws_size < off) return;  // insufficient scratch

  // ---- 1) weights fp32 -> bf16 (once per launch) ----
  cvt_bf16_kernel<<<4096, 256, 0, stream>>>(W1,   W1b,   (long)L_ * (E_ + C_) * H_);
  cvt_gates_kernel<<<4096, 256, 0, stream>>>(Wd, Wf, Wi, Wgb);
  cvt_bf16_kernel<<<4096, 256, 0, stream>>>(Wout, Woutb, (long)H_ * V_);
  pack_bias_kernel<<<(L_ * C_ + 255) / 256, 256, 0, stream>>>(bd, bf_, bi, bcat);

  // ---- 2) init context = 0, embed + LN -> x half of concat buffer ----
  init_ctx_kernel<<<1024, 256, 0, stream>>>(ctxF, fnnB);
  embed_ln_kernel<<<N_, 256, 0, stream>>>(ids, emb, en_g, en_b, fnnB);

  // ---- 3) layer recurrence ----
  for (int l = 0; l < L_; ++l) {
    // hidden = relu([x,ctx] @ W1 + b1) -> bf16
    gemm_bf16_kernel<1, 1><<<dim3(H_ / BN, N_ / BM), 256, 0, stream>>>(
        fnnB, W1b + (size_t)l * (E_ + C_) * H_, b1 + (size_t)l * H_,
        nullptr, hidB, N_, H_, E_ + C_);
    // [delta|forget|ig] = mixedact(hidden @ [Wd|Wf|Wi] + bcat) -> f32 [N,3C]
    gemm_bf16_kernel<4, 0><<<dim3(G3_ / BN, N_ / BM), 256, 0, stream>>>(
        hidB, Wgb + (size_t)l * H_ * G3_, bcat + (size_t)l * G3_,
        gatesF, nullptr, N_, G3_, H_);
    // context = clip(LN(forget*ctx + ig*delta)) -> f32 + bf16 concat half
    ctx_ln_kernel<<<N_, 256, 0, stream>>>(
        gatesF, ctxF, ln_g + (size_t)l * C_, ln_b + (size_t)l * C_, fnnB);
  }

  // ---- 4) logits = hidden @ Wout + bout -> f32 d_out ----
  gemm_bf16_kernel<0, 0><<<dim3(V_ / BN, N_ / BM), 256, 0, stream>>>(
      hidB, Woutb, bout, out, nullptr, N_, V_, H_);
}
